// SddmmLinear_88132728914121
// MI455X (gfx1250) — compile-verified, matrix operand order checked by reading, not run on hardware
//
#include <hip/hip_runtime.h>
#include <hip/hip_bf16.h>

#define IN_F 4096
#define OUT_F 11008
#define NCLUST 16
#define TOPK 1024

typedef __attribute__((ext_vector_type(16))) _Float16 v16h;
typedef __attribute__((ext_vector_type(8)))  float    v8f;

// -------------------------------------------------------------------------
// Kernel 1: softmax(|x|) -> nearest cluster center -> radix-select threshold
// of top-1024 center values -> write masked x (f16) to workspace.
// Single workgroup of 1024 threads; all data is tiny (4096 elems, 16 centers).
// -------------------------------------------------------------------------
__global__ __launch_bounds__(1024) void select_mask_kernel(
    const float* __restrict__ x,
    const float* __restrict__ cc,   // [16, 4096]
    _Float16* __restrict__ xm)      // [4096] out (workspace)
{
    __shared__ float sx[IN_F];
    __shared__ float sact[IN_F];
    __shared__ float red[1024];
    __shared__ unsigned int hist[256];
    __shared__ float d2s[NCLUST];
    __shared__ int   s_ci;
    __shared__ unsigned int s_pref;
    __shared__ int   s_need;

    const int tid = threadIdx.x;

    for (int i = tid; i < IN_F; i += 1024) sx[i] = x[i];
    __syncthreads();

    // softmax(|x|): max
    float m = -1e30f;
    for (int i = tid; i < IN_F; i += 1024) m = fmaxf(m, fabsf(sx[i]));
    red[tid] = m; __syncthreads();
    for (int off = 512; off > 0; off >>= 1) {
        if (tid < off) red[tid] = fmaxf(red[tid], red[tid + off]);
        __syncthreads();
    }
    m = red[0]; __syncthreads();

    // exp + sum
    float sum = 0.0f;
    for (int i = tid; i < IN_F; i += 1024) {
        float e = __expf(fabsf(sx[i]) - m);
        sact[i] = e; sum += e;
    }
    red[tid] = sum; __syncthreads();
    for (int off = 512; off > 0; off >>= 1) {
        if (tid < off) red[tid] += red[tid + off];
        __syncthreads();
    }
    float inv = 1.0f / red[0]; __syncthreads();
    for (int i = tid; i < IN_F; i += 1024) sact[i] *= inv;
    __syncthreads();

    // squared L2 distance to each of the 16 centers
    for (int c = 0; c < NCLUST; ++c) {
        const float* cen = cc + (size_t)c * IN_F;
        float d = 0.0f;
        for (int i = tid; i < IN_F; i += 1024) {
            float t = cen[i] - sact[i];
            d += t * t;
        }
        red[tid] = d; __syncthreads();
        for (int off = 512; off > 0; off >>= 1) {
            if (tid < off) red[tid] += red[tid + off];
            __syncthreads();
        }
        if (tid == 0) d2s[c] = red[0];
        __syncthreads();
    }
    if (tid == 0) {
        int best = 0; float bv = d2s[0];
        for (int c = 1; c < NCLUST; ++c)
            if (d2s[c] < bv) { bv = d2s[c]; best = c; }
        s_ci = best;
    }
    __syncthreads();

    const float* cen = cc + (size_t)s_ci * IN_F;

    // Radix select the TOPK-th largest value. Softmax outputs are positive
    // floats, so IEEE bit patterns compare as unsigned ints.
    unsigned int prefix = 0u;
    int kneed = TOPK;
    for (int level = 0; level < 4; ++level) {
        const int shift = 24 - level * 8;
        if (tid < 256) hist[tid] = 0u;
        __syncthreads();
        const unsigned int pmask =
            (level == 0) ? 0u : (0xFFFFFFFFu << (shift + 8));
        for (int i = tid; i < IN_F; i += 1024) {
            unsigned int key = __float_as_uint(cen[i]);
            if ((key & pmask) == prefix)
                atomicAdd(&hist[(key >> shift) & 0xFFu], 1u);
        }
        __syncthreads();
        if (tid == 0) {
            int cum = 0, b = 255;
            for (; b > 0; --b) {
                if (cum + (int)hist[b] >= kneed) break;
                cum += (int)hist[b];
            }
            s_need = kneed - cum;                      // still needed in bin b
            s_pref = prefix | ((unsigned int)b << shift);
        }
        __syncthreads();
        prefix = s_pref;
        kneed  = s_need;
        __syncthreads();
    }
    const unsigned int T = prefix;   // full 32-bit threshold key

    // Build masked x in f16. Strict >/ < handled in parallel; ties at the
    // threshold picked in ascending-index order (matches lax.top_k).
    for (int i = tid; i < IN_F; i += 1024) {
        unsigned int key = __float_as_uint(cen[i]);
        if (key > T)      xm[i] = (_Float16)sx[i];
        else if (key < T) xm[i] = (_Float16)0.0f;
    }
    __syncthreads();
    if (tid == 0) {
        int need = kneed;
        for (int i = 0; i < IN_F; ++i) {
            if (__float_as_uint(cen[i]) == T) {
                xm[i] = (need > 0) ? (_Float16)sx[i] : (_Float16)0.0f;
                --need;
            }
        }
    }
}

// -------------------------------------------------------------------------
// Kernel 2: out[11008] = xm(4096,f16) . W^T + bias  via v_wmma_f32_16x16x32_f16
// One wave -> 16 outputs (N dim). A row 0 carries the x chunk (rows 1..15 = 0).
// K loops over 4096 in steps of 32. 86 blocks x 8 waves = 688 tiles = 11008.
// -------------------------------------------------------------------------
__global__ __launch_bounds__(256) void gemv_wmma_kernel(
    const _Float16* __restrict__ xm,
    const float* __restrict__ W,     // [OUT_F, IN_F] row-major
    const float* __restrict__ bias,  // [OUT_F]
    float* __restrict__ out)         // [OUT_F]
{
    __shared__ _Float16 sxm[IN_F];   // 8 KB
    {
        const uint32_t* g = (const uint32_t*)xm;
        uint32_t* s = (uint32_t*)sxm;
        for (int i = threadIdx.x; i < IN_F / 2; i += 256) s[i] = g[i];
    }
    __syncthreads();

    const int wave  = threadIdx.x >> 5;
    const int lane  = threadIdx.x & 31;
    const int tile  = blockIdx.x * 8 + wave;
    const int obase = tile * 16;
    if (obase >= OUT_F) return;

    const int n   = lane & 15;            // N column (B) / M row (A)
    const int klo = (lane < 16) ? 0 : 8;  // per-ISA 16-bit operand K striping
    const float* wrow = W + (size_t)(obase + n) * IN_F;
    const bool arow0 = (n == 0);          // lanes 0 and 16 carry A row M=0

    v8f acc = {0.f, 0.f, 0.f, 0.f, 0.f, 0.f, 0.f, 0.f};

    for (int kb = 0; kb < IN_F; kb += 32) {
        __builtin_prefetch(wrow + kb + 128, 0, 1);  // global_prefetch_b8

        // A: 16x32 f16, only row 0 non-zero = xm[kb .. kb+31]
        v16h a = {};
        if (arow0) {
#pragma unroll
            for (int e = 0; e < 8; ++e) {
                a[e]     = sxm[kb + klo + e];          // K = klo+e
                a[e + 8] = sxm[kb + 16 + klo + e];     // K = 16+klo+e
            }
        }

        // B: 32x16 f16 tile of W (K x N), column n, same K striping as A
        const float4 w0 = *(const float4*)(wrow + kb + klo);
        const float4 w1 = *(const float4*)(wrow + kb + klo + 4);
        const float4 w2 = *(const float4*)(wrow + kb + 16 + klo);
        const float4 w3 = *(const float4*)(wrow + kb + 16 + klo + 4);
        v16h b;
        b[0]  = (_Float16)w0.x; b[1]  = (_Float16)w0.y;
        b[2]  = (_Float16)w0.z; b[3]  = (_Float16)w0.w;
        b[4]  = (_Float16)w1.x; b[5]  = (_Float16)w1.y;
        b[6]  = (_Float16)w1.z; b[7]  = (_Float16)w1.w;
        b[8]  = (_Float16)w2.x; b[9]  = (_Float16)w2.y;
        b[10] = (_Float16)w2.z; b[11] = (_Float16)w2.w;
        b[12] = (_Float16)w3.x; b[13] = (_Float16)w3.y;
        b[14] = (_Float16)w3.z; b[15] = (_Float16)w3.w;

        acc = __builtin_amdgcn_wmma_f32_16x16x32_f16(
            false, a, false, b, (short)0, acc, false, false);
    }

    // D layout: VGPR0, lanes 0..15 hold (M=0, N=lane)
    if (lane < 16) {
        out[obase + lane] = acc[0] + bias[obase + lane];
    }
}

// -------------------------------------------------------------------------
extern "C" void kernel_launch(void* const* d_in, const int* in_sizes, int n_in,
                              void* d_out, int out_size, void* d_ws, size_t ws_size,
                              hipStream_t stream) {
    const float* x    = (const float*)d_in[0];
    const float* W    = (const float*)d_in[1];
    const float* bias = (const float*)d_in[2];
    const float* cc   = (const float*)d_in[3];
    float* out        = (float*)d_out;
    _Float16* xm      = (_Float16*)d_ws;   // 4096 f16 = 8 KB scratch

    select_mask_kernel<<<1, 1024, 0, stream>>>(x, cc, xm);

    const int tiles  = OUT_F / 16;              // 688
    const int blocks = (tiles + 7) / 8;         // 86
    gemv_wmma_kernel<<<blocks, 256, 0, stream>>>(xm, W, bias, out);
}
